// BayesianSequenceModel_13761075216825
// MI455X (gfx1250) — matrix operand order repaired and checked by reading, not compiled, not run on hardware
//
#include <hip/hip_runtime.h>

// ---------------------------------------------------------------------------
// BayesianSequenceModel guide: per-step LSTMCell + MLP + reparameterized z
//   N=1024 rows, T=128 steps, A=8, Z=32, H=256
// One persistent kernel, 32 blocks x 256 threads (8 wave32), 32 batch rows
// per block for the whole T loop. All matmuls v_wmma_f32_16x16x32_bf16.
// X activation buffer is ping-ponged so stage-1 fuses WMMA accumulate ->
// LSTM nonlinearity -> h write with no barrier (accumulators and LSTM c
// state live in VGPRs across all 128 steps; no spills).
// MLP weights (112KB) live in dynamic LDS; gate weights (640KB) stream
// from L2 in pre-packed B-fragment layout with prefetch.
// ---------------------------------------------------------------------------

typedef __attribute__((ext_vector_type(16))) __bf16 v16bf;
typedef __attribute__((ext_vector_type(8)))  __bf16 v8bf;
typedef __attribute__((ext_vector_type(8)))  float  v8f;

constexpr int N_   = 1024;
constexpr int T_   = 128;
constexpr int ADIM = 8;
constexpr int ZDIM = 32;
constexpr int HDIM = 256;

constexpr int KG   = 320;   // a(8) + z(32) + h(256) + zero pad(24)
constexpr int NG   = 1024;  // 4*H gate columns
constexpr int NTG  = NG / 16;
constexpr int ROWS = 32;    // batch rows per block
constexpr int LDX  = KG;

constexpr int W1_ELEMS = 256 * 128;
constexpr int W2_ELEMS = 128 * 128;
constexpr int WZ_ELEMS = 128 * 64;
constexpr int WLDS_ELEMS = W1_ELEMS + W2_ELEMS + WZ_ELEMS;  // 57344

// ---------------------------------------------------------------------------
// Weight packing: each (32k x 16n) tile stored as 32 lanes x 16 bf16,
// lane = n' + 16*((k'/8)%2), elem j = k'%8 + 8*(k'/16) -> matches WMMA 16-bit B layout.
// ---------------------------------------------------------------------------
__device__ __forceinline__ int pack_off(int k, int n, int NT) {
  int kt = k >> 5, nt = n >> 4, kk = k & 31, nn = n & 15;
  int lane = nn + (((kk >> 3) & 1) << 4);
  int j = (kk & 7) + ((kk >> 4) << 3);
  return ((kt * NT + nt) << 9) + (lane << 4) + j;
}

__global__ void pack_wg(const float* __restrict__ W_ih,   // [1024,40]
                        const float* __restrict__ W_hh,   // [1024,256]
                        __bf16* __restrict__ dst) {
  int id = blockIdx.x * blockDim.x + threadIdx.x;
  if (id >= KG * NG) return;
  int k = id / NG, n = id % NG;
  float v = 0.f;
  if (k < 40)        v = W_ih[n * 40 + k];          // x = [a(8) | z(32)]
  else if (k < 296)  v = W_hh[n * 256 + (k - 40)];  // h
  dst[pack_off(k, n, NTG)] = (__bf16)v;
}

__global__ void pack_w(const float* __restrict__ W,  // [Nw, Kw] row-major
                       __bf16* __restrict__ dst, int Nw, int Kw) {
  int id = blockIdx.x * blockDim.x + threadIdx.x;
  if (id >= Nw * Kw) return;
  int k = id / Nw, n = id % Nw;
  dst[pack_off(k, n, Nw >> 4)] = (__bf16)W[n * Kw + k];
}

__global__ void pack_bias(const float* __restrict__ b_ih,
                          const float* __restrict__ b_hh,
                          float* __restrict__ bg) {
  int id = blockIdx.x * blockDim.x + threadIdx.x;
  if (id < NG) bg[id] = b_ih[id] + b_hh[id];
}

// ---------------------------------------------------------------------------
// Fragment loaders
// ---------------------------------------------------------------------------
__device__ __forceinline__ v16bf load_bfrag(const __bf16* __restrict__ Wp,
                                            int NT, int kt, int nt, int lane) {
  const __bf16* p = Wp + (((kt * NT + nt) << 9) + (lane << 4));
  v8bf lo = *(const v8bf*)(p);
  v8bf hi = *(const v8bf*)(p + 8);
  return __builtin_shufflevector(lo, hi, 0, 1, 2, 3, 4, 5, 6, 7,
                                 8, 9, 10, 11, 12, 13, 14, 15);
}

// 16-bit A 16x32 layout: lanes 0-15 hold M, K kb..kb+7 / kb+16..kb+23;
// lanes 16-31 the +8 shifted halves (kb pre-offset with khalf*8).
__device__ __forceinline__ v16bf load_afrag(const __bf16* __restrict__ X,
                                            int ld, int M, int kb) {
  const __bf16* p = X + M * ld + kb;
  v8bf lo = *(const v8bf*)(p);
  v8bf hi = *(const v8bf*)(p + 16);
  return __builtin_shufflevector(lo, hi, 0, 1, 2, 3, 4, 5, 6, 7,
                                 8, 9, 10, 11, 12, 13, 14, 15);
}

#define WMMA_BF16(a, b, c)                                                  \
  __builtin_amdgcn_wmma_f32_16x16x32_bf16(false, (a), false, (b), (short)0, \
                                          (c), false, false)

__device__ __forceinline__ float sigmoid_(float x) {
  return 1.f / (1.f + __expf(-x));
}
__device__ __forceinline__ float tanh_(float x) {
  // stable: tanh(|x|) then restore sign (avoids inf/inf NaN for x << 0)
  float e = __expf(-2.f * fabsf(x));
  float r = (1.f - e) / (1.f + e);
  return copysignf(r, x);
}
__device__ __forceinline__ float softplus_(float x) {
  return (x > 20.f) ? x : __logf(1.f + __expf(x));
}

// ---------------------------------------------------------------------------
// Main persistent sequence kernel
// ---------------------------------------------------------------------------
__global__ void __launch_bounds__(256)
seq_kernel(const float* __restrict__ Ain,   // [N,T,8]
           const float* __restrict__ epsin, // [N,T,32]
           const float* __restrict__ z0, const float* __restrict__ h0,
           const float* __restrict__ c0,
           const __bf16* __restrict__ Wg,   // packed [320x1024]
           const float* __restrict__ bg,    // [1024] combined gate bias
           const __bf16* __restrict__ W1p, const float* __restrict__ b1,
           const __bf16* __restrict__ W2p, const float* __restrict__ b2,
           const __bf16* __restrict__ Wzp, const float* __restrict__ bz,
           float* __restrict__ Zout) {      // [N,T,32]
  // ping-pong activation buffer [a|z|h|pad] + MLP activations (static LDS)
  __shared__ __attribute__((aligned(16))) __bf16 Xs[2][ROWS * KG];
  __shared__ __attribute__((aligned(16))) __bf16 U1[ROWS * 128];
  __shared__ __attribute__((aligned(16))) __bf16 U2[ROWS * 128];
  // MLP weights resident in dynamic LDS for the whole kernel
  extern __shared__ __attribute__((aligned(16))) __bf16 wlds[];
  __bf16* W1L = wlds;                 // 256x128 packed
  __bf16* W2L = W1L + W1_ELEMS;       // 128x128 packed
  __bf16* WzL = W2L + W2_ELEMS;       // 128x64  packed

  const int tid   = threadIdx.x;
  const int wave  = tid >> 5;
  const int lane  = tid & 31;
  const int lrow  = lane & 15;
  const int khalf = (lane >> 4) & 1;
  const int row0  = blockIdx.x * ROWS;

  // ---- copy MLP weights into LDS (16B chunks) ----
  for (int c = tid; c < WLDS_ELEMS / 8; c += 256) {
    const __bf16* src = (c < W1_ELEMS / 8)        ? W1p + c * 8
                        : (c < (W1_ELEMS + W2_ELEMS) / 8)
                            ? W2p + (c - W1_ELEMS / 8) * 8
                            : Wzp + (c - (W1_ELEMS + W2_ELEMS) / 8) * 8;
    *(v8bf*)(wlds + c * 8) = *(const v8bf*)src;
  }

  // ---- init X[0] (z0, h0) and pads in both buffers ----
  for (int idx = tid; idx < ROWS * ZDIM; idx += 256)
    Xs[0][(idx >> 5) * LDX + ADIM + (idx & 31)] = (__bf16)z0[idx & 31];
  for (int idx = tid; idx < ROWS * HDIM; idx += 256)
    Xs[0][(idx >> 8) * LDX + 40 + (idx & 255)] = (__bf16)h0[idx & 255];
  for (int idx = tid; idx < ROWS * 24; idx += 256) {
    int r = idx / 24, k = idx % 24;
    Xs[0][r * LDX + 296 + k] = (__bf16)0.f;
    Xs[1][r * LDX + 296 + k] = (__bf16)0.f;
  }

  // ---- t-invariant per-lane state: c fragments + biases (hoisted) ----
  v8f cst[4];
  float bI[4], bF[4], bG[4], bO[4];
#pragma unroll
  for (int i = 0; i < 4; ++i) {
    int ct = (wave * 4 + i) & 15;
    int col = ct * 16 + lrow;
    float cv = c0[col];  // c0 broadcast over rows
#pragma unroll
    for (int r = 0; r < 8; ++r) cst[i][r] = cv;
    bI[i] = bg[col];
    bF[i] = bg[col + 256];
    bG[i] = bg[col + 512];
    bO[i] = bg[col + 768];
  }
  const float b1v = b1[wave * 16 + lrow];
  const float b2v = b2[wave * 16 + lrow];
  const int jz = (wave & 1) * 16 + lrow;  // stage-4 z column (waves 0-3)
  const float bzL = bz[jz & 31];
  const float bzR = bz[ZDIM + (jz & 31)];
  __syncthreads();

  for (int t = 0; t < T_; ++t) {
    const int cur = t & 1;
    __bf16* Xc = Xs[cur];       // read:  a_t, z_t, h_t
    __bf16* Xn = Xs[cur ^ 1];   // write: h_{t+1}, z_{t+1}

    // ---- stream a_t into X[cur] ----
    {
      int rr = tid >> 3, k = tid & 7;  // 256 threads cover 32x8
      Xc[rr * LDX + k] =
          (__bf16)Ain[((size_t)(row0 + rr) * T_ + t) * ADIM + k];
    }
    __syncthreads();  // a_t + prev-step z/h writes all visible

    // ---- Stage 1: gates, LSTM nonlinearity, h -> X[next] (fused) ----
#pragma unroll
    for (int i = 0; i < 4; ++i) {
      int id = wave * 4 + i;
      int mt = id >> 4, ct = id & 15;
      int M = mt * 16 + lrow;
      v8f ci = {}, cf = {}, cg = {}, co = {};
#pragma unroll 1
      for (int kt = 0; kt < KG / 32; ++kt) {
        // unconditional: kt=9 prefetch lands in adjacent packed-W1 region
        __builtin_prefetch(Wg + (((kt + 1) * NTG + ct) << 9) + (lane << 4),
                           0, 1);
        v16bf a   = load_afrag(Xc, LDX, M, kt * 32 + khalf * 8);
        v16bf wbi = load_bfrag(Wg, NTG, kt, ct,      lane);
        v16bf wbf = load_bfrag(Wg, NTG, kt, ct + 16, lane);
        v16bf wbg = load_bfrag(Wg, NTG, kt, ct + 32, lane);
        v16bf wbo = load_bfrag(Wg, NTG, kt, ct + 48, lane);
        ci = WMMA_BF16(a, wbi, ci);
        cf = WMMA_BF16(a, wbf, cf);
        cg = WMMA_BF16(a, wbg, cg);
        co = WMMA_BF16(a, wbo, co);
      }
      int col = ct * 16 + lrow;
      int mbase = mt * 16 + khalf * 8;
#pragma unroll
      for (int r = 0; r < 8; ++r) {
        float gi = sigmoid_(ci[r] + bI[i]);
        float gf = sigmoid_(cf[r] + bF[i]);
        float gg = tanh_(cg[r] + bG[i]);
        float go = sigmoid_(co[r] + bO[i]);
        float cn = gf * cst[i][r] + gi * gg;
        cst[i][r] = cn;
        Xn[(mbase + r) * LDX + 40 + col] = (__bf16)(go * tanh_(cn));
      }
    }
    __syncthreads();  // h_{t+1} visible in X[next]

    // ---- Stage 2: u1 = relu(h_new @ W1^T + b1) ----
    {
      int nt = wave;
#pragma unroll
      for (int mt = 0; mt < 2; ++mt) {
        v8f acc = {};
        int M = mt * 16 + lrow;
#pragma unroll 1
        for (int kt = 0; kt < 8; ++kt) {
          v16bf a = load_afrag(Xn, LDX, M, 40 + kt * 32 + khalf * 8);
          v16bf b = load_bfrag(W1L, 8, kt, nt, lane);
          acc = WMMA_BF16(a, b, acc);
        }
        int mbase = mt * 16 + khalf * 8;
#pragma unroll
        for (int r = 0; r < 8; ++r)
          U1[(mbase + r) * 128 + nt * 16 + lrow] =
              (__bf16)fmaxf(acc[r] + b1v, 0.f);
      }
    }
    __syncthreads();

    // ---- Stage 3: u2 = relu(u1 @ W2^T + b2) ----
    {
      int nt = wave;
#pragma unroll
      for (int mt = 0; mt < 2; ++mt) {
        v8f acc = {};
        int M = mt * 16 + lrow;
#pragma unroll 1
        for (int kt = 0; kt < 4; ++kt) {
          v16bf a = load_afrag(U1, 128, M, kt * 32 + khalf * 8);
          v16bf b = load_bfrag(W2L, 8, kt, nt, lane);
          acc = WMMA_BF16(a, b, acc);
        }
        int mbase = mt * 16 + khalf * 8;
#pragma unroll
        for (int r = 0; r < 8; ++r)
          U2[(mbase + r) * 128 + nt * 16 + lrow] =
              (__bf16)fmaxf(acc[r] + b2v, 0.f);
      }
    }
    __syncthreads();

    // ---- Stage 4: zz = u2 @ Wz^T; z = loc + softplus(raw)*eps ----
    if (wave < 4) {
      int mt = wave >> 1, ntz = wave & 1;
      v8f al = {}, ar = {};
      int M = mt * 16 + lrow;
#pragma unroll 1
      for (int kt = 0; kt < 4; ++kt) {
        v16bf a  = load_afrag(U2, 128, M, kt * 32 + khalf * 8);
        v16bf bl = load_bfrag(WzL, 4, kt, ntz,     lane);
        v16bf br = load_bfrag(WzL, 4, kt, ntz + 2, lane);
        al = WMMA_BF16(a, bl, al);
        ar = WMMA_BF16(a, br, ar);
      }
      int mbase = mt * 16 + khalf * 8;
#pragma unroll
      for (int r = 0; r < 8; ++r) {
        int rl = mbase + r;
        size_t grow = (size_t)(row0 + rl);
        float loc = al[r] + bzL;
        float sp  = softplus_(ar[r] + bzR);
        float e   = epsin[(grow * T_ + t) * ZDIM + jz];
        float zv  = loc + sp * e;
        Zout[(grow * T_ + t) * ZDIM + jz] = zv;
        Xn[rl * LDX + ADIM + jz] = (__bf16)zv;  // feeds z_{t+1}
      }
    }
    // loop-top barrier orders X[next] z writes vs next step reads
  }
}

// ---------------------------------------------------------------------------
// Launch
// ---------------------------------------------------------------------------
extern "C" void kernel_launch(void* const* d_in, const int* in_sizes, int n_in,
                              void* d_out, int out_size, void* d_ws,
                              size_t ws_size, hipStream_t stream) {
  (void)in_sizes; (void)n_in; (void)out_size; (void)ws_size;
  const float* A    = (const float*)d_in[0];
  const float* eps  = (const float*)d_in[1];
  const float* z0   = (const float*)d_in[2];
  const float* h0   = (const float*)d_in[3];
  const float* c0   = (const float*)d_in[4];
  const float* W_ih = (const float*)d_in[5];
  const float* W_hh = (const float*)d_in[6];
  const float* b_ih = (const float*)d_in[7];
  const float* b_hh = (const float*)d_in[8];
  const float* W1   = (const float*)d_in[9];
  const float* b1   = (const float*)d_in[10];
  const float* W2   = (const float*)d_in[11];
  const float* b2   = (const float*)d_in[12];
  const float* Wz   = (const float*)d_in[13];
  const float* bz   = (const float*)d_in[14];

  __bf16* Wg  = (__bf16*)d_ws;                  // 320*1024 packed bf16
  __bf16* W1p = Wg  + KG * NG;
  __bf16* W2p = W1p + W1_ELEMS;
  __bf16* Wzp = W2p + W2_ELEMS;
  float*  bg  = (float*)(Wzp + WZ_ELEMS);       // 1024 f32

  pack_wg<<<(KG * NG + 255) / 256, 256, 0, stream>>>(W_ih, W_hh, Wg);
  pack_w<<<(W1_ELEMS + 255) / 256, 256, 0, stream>>>(W1, W1p, 128, 256);
  pack_w<<<(W2_ELEMS + 255) / 256, 256, 0, stream>>>(W2, W2p, 128, 128);
  pack_w<<<(WZ_ELEMS + 255) / 256, 256, 0, stream>>>(Wz, Wzp, 64, 128);
  pack_bias<<<(NG + 255) / 256, 256, 0, stream>>>(b_ih, b_hh, bg);

  seq_kernel<<<N_ / ROWS, 256, WLDS_ELEMS * (int)sizeof(__bf16), stream>>>(
      A, eps, z0, h0, c0, Wg, bg, W1p, b1, W2p, b2, Wzp, bz, (float*)d_out);
}
